// S_GCN_65000035057718
// MI455X (gfx1250) — compile-verified, matrix operand order checked by reading, not run on hardware
//
#include <hip/hip_runtime.h>

// Problem constants from the reference
#define B_  8
#define C_  32
#define N_  4096            // H*W = 64*64

typedef __attribute__((ext_vector_type(16))) _Float16 v16h;
typedef __attribute__((ext_vector_type(8)))  _Float16 v8h;
typedef __attribute__((ext_vector_type(8)))  float    v8f;

// -------- Kernel 1: c_adj[b,n] = mean over channels of x --------------------
__global__ __launch_bounds__(256)
void channel_mean_kernel(const float* __restrict__ x, float* __restrict__ cadj) {
    int i = blockIdx.x * blockDim.x + threadIdx.x;     // i in [0, B*N)
    if (i >= B_ * N_) return;
    int b = i >> 12;                                    // / N_
    int n = i & (N_ - 1);
    const float* p = x + (size_t)b * C_ * N_ + n;
    float s = 0.0f;
#pragma unroll
    for (int c = 0; c < C_; ++c) s += p[(size_t)c * N_];
    cadj[i] = s * (1.0f / 32.0f);
}

// -------- Kernel 2: fused A-construction + WMMA GEMM + para*relu ------------
// grid = (N/64, B), block = 128 threads (4 wave32). Each block computes
// out[b, 0:32, m0:m0+64]. Each wave owns a 16-wide m sub-tile with two
// v_wmma_f32_16x16x32_f16 accumulators (c rows 0-15 and 16-31).
// Double-buffered LDS: stage tile i+1 while WMMAs consume tile i.
__global__ __launch_bounds__(128)
void fused_gcn_wmma_kernel(const float* __restrict__ x,
                           const float* __restrict__ para,
                           const float* __restrict__ adj,
                           const float* __restrict__ cadj,
                           float* __restrict__ out) {
    constexpr int MT  = 64;   // m-tile per block
    constexpr int KT  = 32;   // K step (WMMA K)
    constexpr int LDF = 40;   // LDS row stride in halfs: 80B, 16B-aligned

    __shared__ alignas(16) _Float16 sFea[2][C_ * LDF]; // fea tile [c=32][k=32]
    __shared__ alignas(16) _Float16 sA  [2][MT * LDF]; // A^T tile [m=64][k=32]

    const int tid  = threadIdx.x;
    const int wave = tid >> 5;
    const int lane = tid & 31;
    const int b    = blockIdx.y;
    const int m0   = blockIdx.x * MT;

    const float* xb = x    + (size_t)b * C_ * N_;
    const float* cb = cadj + (size_t)b * N_;

    // staging geometry (uniform per thread across iterations)
    const int fc  = tid >> 2;          // fea row (c)        0..31
    const int fks = (tid & 3) << 3;    // fea k-segment      {0,8,16,24}
    const int am  = tid & 63;          // A column (m)       0..63
    const int aks = (tid >> 6) << 4;   // A k-segment        {0,16}
    const float cmv = cb[m0 + am];     // c_adj at this thread's m (loop-invariant)

    // Fragment gather geometry (16-bit A/B operand layout, wave32):
    //   lanes 0-15 : elements 0..7 = K 0..7,  elements 8..15 = K 16..23
    //   lanes 16-31: elements 0..7 = K 8..15, elements 8..15 = K 24..31
    const int kBase = (lane & 16) ? 8 : 0;
    const int row16 = lane & 15;

    v8f acc0 = {};   // c rows 0..15
    v8f acc1 = {};   // c rows 16..31

    // ---- tile stager: batched loads first, then convert + vector LDS store
    auto stage = [&](int buf, int n0) {
        // fea tile: 2x global_load_b128, 1x ds_store_b128 per thread
        const float4* fs =
            reinterpret_cast<const float4*>(xb + (size_t)fc * N_ + n0 + fks);
        float4 f0 = fs[0];
        float4 f1 = fs[1];

        // A tile: 16 independent coalesced b32 loads (one adj row-segment per
        // instruction across the wave), all issued before first use.
        float av[16];
#pragma unroll
        for (int t = 0; t < 16; ++t)
            av[t] = adj[(size_t)(n0 + aks + t) * N_ + (m0 + am)];
        float cn[16];
#pragma unroll
        for (int t = 0; t < 16; ++t)
            cn[t] = cb[n0 + aks + t];

        v8h hf;
        hf[0] = (_Float16)f0.x; hf[1] = (_Float16)f0.y;
        hf[2] = (_Float16)f0.z; hf[3] = (_Float16)f0.w;
        hf[4] = (_Float16)f1.x; hf[5] = (_Float16)f1.y;
        hf[6] = (_Float16)f1.z; hf[7] = (_Float16)f1.w;
        *reinterpret_cast<v8h*>(&sFea[buf][fc * LDF + fks]) = hf;

        // A[n,m] = adj[n,m] * 2/(1+exp(|cadj[m]-cadj[n]|))  (symmetrization is
        // the identity: the weight is even in the difference).
        v8h h0, h1;
#pragma unroll
        for (int t = 0; t < 16; ++t) {
            float d = fabsf(cmv - cn[t]);
            float w = 2.0f / (1.0f + __expf(d));   // v_exp_f32 transcendental
            _Float16 a = (_Float16)(av[t] * w);
            if (t < 8) h0[t] = a; else h1[t - 8] = a;
        }
        *reinterpret_cast<v8h*>(&sA[buf][am * LDF + aks])     = h0;
        *reinterpret_cast<v8h*>(&sA[buf][am * LDF + aks + 8]) = h1;
    };

    // ---- software pipeline: prologue stage, then stage(i+1) || wmma(i) -----
    stage(0, 0);
    __syncthreads();

    int buf = 0;
    for (int n0 = 0; n0 < N_; n0 += KT) {
        const int nn = n0 + KT;
        if (nn < N_) stage(buf ^ 1, nn);    // uniform branch, no lane divergence

        // gather fragments (two ds_load_b128 per operand)
        const _Float16* fb0 = &sFea[buf][(row16     ) * LDF + kBase];
        const _Float16* fb1 = &sFea[buf][(row16 + 16) * LDF + kBase];
        const _Float16* bb  = &sA  [buf][(wave * 16 + row16) * LDF + kBase];

        v16h af0, af1, bfv;
        ((v8h*)&af0)[0] = *(const v8h*)(fb0);
        ((v8h*)&af0)[1] = *(const v8h*)(fb0 + 16);
        ((v8h*)&af1)[0] = *(const v8h*)(fb1);
        ((v8h*)&af1)[1] = *(const v8h*)(fb1 + 16);
        ((v8h*)&bfv)[0] = *(const v8h*)(bb);
        ((v8h*)&bfv)[1] = *(const v8h*)(bb  + 16);

        // D = A(16x32 f16: c x n) * B(32x16 f16: n x m) + C(f32)
        acc0 = __builtin_amdgcn_wmma_f32_16x16x32_f16(
                   false, af0, false, bfv, (short)0, acc0, false, false);
        acc1 = __builtin_amdgcn_wmma_f32_16x16x32_f16(
                   false, af1, false, bfv, (short)0, acc1, false, false);

        __syncthreads();   // frag reads of `buf` done; next iter may overwrite
        buf ^= 1;
    }

    // --- epilogue: para scale + relu per 16x16 f32 C/D layout ---------------
    // lane 0-15: N=lane, VGPR r -> M=r ; lane 16-31: N=lane-16, M=r+8
    const int col    = m0 + wave * 16 + row16;
    const int rowOff = (lane >> 4) * 8;
    float* ob = out + (size_t)b * C_ * N_;
#pragma unroll
    for (int r = 0; r < 8; ++r) {
        int c0 = r + rowOff;                 // 0..15
        float v0 = acc0[r] * para[(size_t)c0 * N_ + col];
        ob[(size_t)c0 * N_ + col] = fmaxf(v0, 0.0f);
        int c1 = c0 + 16;                    // 16..31
        float v1 = acc1[r] * para[(size_t)c1 * N_ + col];
        ob[(size_t)c1 * N_ + col] = fmaxf(v1, 0.0f);
    }
}

// ---------------------------------------------------------------------------
extern "C" void kernel_launch(void* const* d_in, const int* in_sizes, int n_in,
                              void* d_out, int out_size, void* d_ws, size_t ws_size,
                              hipStream_t stream) {
    const float* x    = (const float*)d_in[0];   // [8,32,64,64]
    const float* para = (const float*)d_in[1];   // [1,32,64,64]
    const float* adj  = (const float*)d_in[2];   // [4096,4096]
    float*       out  = (float*)d_out;           // [8,32,64,64]
    float*       cadj = (float*)d_ws;            // [8,4096] = 128 KB scratch

    // 1) channel mean
    {
        int total = B_ * N_;
        channel_mean_kernel<<<(total + 255) / 256, 256, 0, stream>>>(x, cadj);
    }
    // 2) fused adjacency-weighting + WMMA GEMM + para*relu
    {
        dim3 grid(N_ / 64, B_);
        fused_gcn_wmma_kernel<<<grid, 128, 0, stream>>>(x, para, adj, cadj, out);
    }
}
